// HyenaFilter_15547781612206
// MI455X (gfx1250) — matrix operand order repaired and checked by reading, not compiled, not run on hardware
//
#include <hip/hip_runtime.h>
#include <hip/hip_bf16.h>

#define LSEQ 8192
#define NFFT 16384
#define DDIM 768
#define BDIM 4
#define FO   64

typedef __attribute__((ext_vector_type(16))) _Float16 v16h;
typedef __attribute__((ext_vector_type(8)))  float    v8f;

extern __shared__ float smem[];

// Truncate a flat pointer to a __shared__ object to its 32-bit LDS address
// (ISA 10.2: LDS aperture maps addr[31:0] to the wave-relative LDS address).
__device__ __forceinline__ unsigned lds_addr32(const void* p)
{
    return (unsigned)(size_t)p;
}

// ---------------------------------------------------------------------------
// Kernel 1: implicit filter MLP with WMMA (f16 in, f32 acc) -> kT [D][L]
// One wave (32 threads) per 16-row tile of L.
// ---------------------------------------------------------------------------
__global__ __launch_bounds__(32) void filter_kernel(
    const float* __restrict__ w0, const float* __restrict__ b0,
    const float* __restrict__ w1, const float* __restrict__ b1,
    const float* __restrict__ w2, const float* __restrict__ b2,
    const float* __restrict__ freq, const float* __restrict__ deltas,
    float* __restrict__ kT)
{
    __shared__ _Float16 Ah[2][16][FO];   // staging for A fragments

    const int l  = threadIdx.x;          // lane 0..31
    const int m0 = blockIdx.x * 16;      // row-tile base in L

    // ---- stage 1: h0 = sin(freq * (z @ w0 + b0)), z = [t, cos, -sin] ----
    for (int idx = l; idx < 16 * FO; idx += 32) {
        int m = idx >> 6, j = idx & 63;
        float mm = (float)(m0 + m);
        float t  = mm * (1.0f / (float)(LSEQ - 1));
        float w  = 6.283185307179586f * mm * (1.0f / (float)LSEQ);
        float s, c;
        __sincosf(1e-4f * w, &s, &c);
        float acc = t * w0[j] + c * w0[FO + j] + (-s) * w0[2 * FO + j] + b0[j];
        Ah[0][m][j] = (_Float16)__sinf(freq[j] * acc);
    }
    __syncthreads();

    const int row = l & 15;          // A: M row / B: N col / C: N col
    const int kb  = (l >> 4) * 8;    // A-fragment K sub-base
    const int kgb = (l >> 4) * 16;   // unused; B fragment indexes from kb like A

    // ---- stage 2: h1 = sin(freq * (h0 @ w1 + b1)) : 4 N-tiles x 2 K-chunks ----
    for (int nt = 0; nt < FO / 16; ++nt) {
        int n0 = nt * 16;
        v8f cacc = {};
        for (int kt = 0; kt < 2; ++kt) {
            int k0 = kt * 32;
            v16h a, b;
            for (int j = 0; j < 8; ++j) {
                a[j]     = Ah[0][row][k0 + kb + j];
                a[8 + j] = Ah[0][row][k0 + 16 + kb + j];
            }
            for (int e = 0; e < 16; ++e)
                b[e] = (_Float16)w1[(k0 + kb + (e & 7) + ((e >> 3) << 4)) * FO + (n0 + row)];
            cacc = __builtin_amdgcn_wmma_f32_16x16x32_f16(
                false, a, false, b, (short)0, cacc, false, false);
        }
        int n = n0 + row;
        for (int i = 0; i < 8; ++i) {
            int m = i + (l >> 4) * 8;
            Ah[1][m][n] = (_Float16)__sinf(freq[n] * (cacc[i] + b1[n]));
        }
    }
    __syncthreads();

    // ---- stage 3: k = (h1 @ w2 + b2) * (exp(-t*|delta|) + SHIFT) ----
    for (int nt = 0; nt < DDIM / 16; ++nt) {
        int n0 = nt * 16;
        v8f cacc = {};
        for (int kt = 0; kt < 2; ++kt) {
            int k0 = kt * 32;
            v16h a, b;
            for (int j = 0; j < 8; ++j) {
                a[j]     = Ah[1][row][k0 + kb + j];
                a[8 + j] = Ah[1][row][k0 + 16 + kb + j];
            }
            for (int e = 0; e < 16; ++e)
                b[e] = (_Float16)w2[(k0 + kb + (e & 7) + ((e >> 3) << 4)) * DDIM + (n0 + row)];
            cacc = __builtin_amdgcn_wmma_f32_16x16x32_f16(
                false, a, false, b, (short)0, cacc, false, false);
        }
        int n = n0 + row;
        float ad = fabsf(deltas[n]);
        for (int i = 0; i < 8; ++i) {
            int m = m0 + i + (l >> 4) * 8;
            float t = (float)m * (1.0f / (float)(LSEQ - 1));
            float dec = __expf(-t * ad);              // SHIFT == 0.0
            kT[(size_t)n * LSEQ + m] = (cacc[i] + b2[n]) * dec;
        }
    }
}

// ---------------------------------------------------------------------------
// In-LDS radix-2 FFT (256 threads, N = 16384) with a shared twiddle table:
//   tw[k] = (cos(2*pi*k/N), sin(2*pi*k/N)), k < N/2.
// Stage s (size = 2^s) uses tw[j << (14-s)]; DIF takes the conjugate.
// fft_dif : natural order in -> bit-reversed out   (e^{-i...})
// fft_dit : bit-reversed in -> natural order out   (e^{+i...})
// ---------------------------------------------------------------------------
__device__ __forceinline__ void build_twiddles(float2* tw, int tid)
{
    for (int k = tid; k < NFFT / 2; k += 256) {
        float s, c;
        __sincosf(6.283185307179586f * (float)k * (1.0f / (float)NFFT), &s, &c);
        tw[k] = make_float2(c, s);
    }
}

__device__ __forceinline__ void fft_dif(float* re, float* im,
                                        const float2* tw, int tid)
{
    for (int s = 14; s >= 1; --s) {
        int half = 1 << (s - 1);
        int shift = 14 - s;
        for (int idx = tid; idx < NFFT / 2; idx += 256) {
            int j    = idx & (half - 1);
            int p0   = ((idx >> (s - 1)) << s) + j;
            int p1   = p0 + half;
            float2 w = tw[j << shift];
            float tr = w.x, ti = -w.y;                 // conjugate
            float ur = re[p0], ui = im[p0];
            float vr = re[p1], vi = im[p1];
            float dr = ur - vr, di = ui - vi;
            re[p0] = ur + vr;           im[p0] = ui + vi;
            re[p1] = dr * tr - di * ti; im[p1] = dr * ti + di * tr;
        }
        __syncthreads();
    }
}

__device__ __forceinline__ void fft_dit(float* re, float* im,
                                        const float2* tw, int tid)
{
    for (int s = 1; s <= 14; ++s) {
        int half = 1 << (s - 1);
        int shift = 14 - s;
        for (int idx = tid; idx < NFFT / 2; idx += 256) {
            int j    = idx & (half - 1);
            int p0   = ((idx >> (s - 1)) << s) + j;
            int p1   = p0 + half;
            float2 w = tw[j << shift];
            float tr = w.x, ti = w.y;
            float vr = re[p1] * tr - im[p1] * ti;
            float vi = re[p1] * ti + im[p1] * tr;
            float ur = re[p0], ui = im[p0];
            re[p0] = ur + vr; im[p0] = ui + vi;
            re[p1] = ur - vr; im[p1] = ui - vi;
        }
        __syncthreads();
    }
}

// Async bulk copy: 32KB (LSEQ floats) global -> LDS, ASYNCcnt-tracked.
__device__ __forceinline__ void async_load_seq(float* re, const float* gsrc, int tid)
{
    unsigned base = lds_addr32(re);
    for (int pass = 0; pass < LSEQ / (256 * 4); ++pass) {
        unsigned off = (unsigned)(pass * 256 + tid) * 16u;
        asm volatile("global_load_async_to_lds_b128 %0, %1, %2"
                     :: "v"(base + off), "v"(off), "s"(gsrc)
                     : "memory");
    }
}

// Async bulk copy: 32KB LDS -> global, ASYNCcnt-tracked.
__device__ __forceinline__ void async_store_seq(const float* re, float* gdst, int tid)
{
    unsigned base = lds_addr32(re);
    for (int pass = 0; pass < LSEQ / (256 * 4); ++pass) {
        unsigned off = (unsigned)(pass * 256 + tid) * 16u;
        asm volatile("global_store_async_from_lds_b128 %0, %1, %2"
                     :: "v"(off), "v"(base + off), "s"(gdst)
                     : "memory");
    }
}

__device__ __forceinline__ void wait_async0()
{
    asm volatile("s_wait_asynccnt 0x0" ::: "memory");
}

// ---------------------------------------------------------------------------
// Kernel 2: per-channel forward FFT of filter -> scrambled spectrum * (1/N)
// ---------------------------------------------------------------------------
__global__ __launch_bounds__(256) void kfft_kernel(
    const float* __restrict__ kT, float* __restrict__ Kf)
{
    float*  re = smem;
    float*  im = smem + NFFT;
    float2* tw = (float2*)(smem + 2 * NFFT);
    int d = blockIdx.x, tid = threadIdx.x;

    async_load_seq(re, kT + (size_t)d * LSEQ, tid);     // re[0..L-1] async
    build_twiddles(tw, tid);
    for (int i = tid; i < NFFT; i += 256) {
        if (i >= LSEQ) re[i] = 0.0f;
        im[i] = 0.0f;
    }
    wait_async0();
    __syncthreads();

    fft_dif(re, im, tw, tid);

    float2* dst = (float2*)Kf + (size_t)d * NFFT;
    const float inv = 1.0f / (float)NFFT;               // rfft(k)/fft_size
    for (int i = tid; i < NFFT; i += 256)
        dst[i] = make_float2(re[i] * inv, im[i] * inv);
}

// ---------------------------------------------------------------------------
// Kernel 3: transpose x [B,L,D] -> xT [B,D,L] (coalesced via LDS tiles)
// ---------------------------------------------------------------------------
__global__ __launch_bounds__(256) void transpose_fwd(
    const float* __restrict__ x, float* __restrict__ xT)
{
    __shared__ float tile[32][33];
    int b  = blockIdx.z;
    int l0 = blockIdx.x * 32, d0 = blockIdx.y * 32;
    int tx = threadIdx.x, ty = threadIdx.y;
    for (int r = ty; r < 32; r += 8)
        tile[r][tx] = x[(size_t)b * LSEQ * DDIM + (size_t)(l0 + r) * DDIM + (d0 + tx)];
    __syncthreads();
    for (int r = ty; r < 32; r += 8)
        xT[((size_t)b * DDIM + (d0 + r)) * LSEQ + (l0 + tx)] = tile[tx][r];
}

// ---------------------------------------------------------------------------
// Kernel 4: fused FFT conv per (b,d): y = IDFT(DFT(x) .* Kf)/N + bias
// Pointwise product in bit-reversed order (DIF -> DIT, no reversal pass).
// Result written back over xT in place via async LDS->global store.
// ---------------------------------------------------------------------------
__global__ __launch_bounds__(256) void conv_kernel(
    float* __restrict__ xT, const float* __restrict__ Kf,
    const float* __restrict__ bias)
{
    float*  re = smem;
    float*  im = smem + NFFT;
    float2* tw = (float2*)(smem + 2 * NFFT);
    int seq = blockIdx.x;            // b*D + d
    int d   = seq % DDIM;
    int tid = threadIdx.x;
    float* x = xT + (size_t)seq * LSEQ;

    async_load_seq(re, x, tid);                         // re[0..L-1] async
    build_twiddles(tw, tid);
    for (int i = tid; i < NFFT; i += 256) {
        if (i >= LSEQ) re[i] = 0.0f;
        im[i] = 0.0f;
    }
    wait_async0();
    __syncthreads();

    fft_dif(re, im, tw, tid);

    const float2* kf = (const float2*)Kf + (size_t)d * NFFT;
    for (int i = tid; i < NFFT; i += 256) {
        float2 kv = kf[i];
        float ur = re[i], ui = im[i];
        re[i] = ur * kv.x - ui * kv.y;
        im[i] = ur * kv.y + ui * kv.x;
    }
    __syncthreads();

    fft_dit(re, im, tw, tid);

    const float inv = 1.0f / (float)NFFT;               // irfft backward norm
    float bd = bias[d];
    for (int i = tid; i < LSEQ; i += 256)
        re[i] = re[i] * inv + bd;                       // finalize in LDS
    __syncthreads();

    async_store_seq(re, x, tid);                        // LDS -> global async
    wait_async0();
}

// ---------------------------------------------------------------------------
// Kernel 5: transpose back xT [B,D,L] -> out [B,L,D]
// ---------------------------------------------------------------------------
__global__ __launch_bounds__(256) void transpose_bwd(
    const float* __restrict__ xT, float* __restrict__ out)
{
    __shared__ float tile[32][33];
    int b  = blockIdx.z;
    int l0 = blockIdx.x * 32, d0 = blockIdx.y * 32;
    int tx = threadIdx.x, ty = threadIdx.y;
    for (int r = ty; r < 32; r += 8)
        tile[r][tx] = xT[((size_t)b * DDIM + (d0 + r)) * LSEQ + (l0 + tx)];
    __syncthreads();
    for (int r = ty; r < 32; r += 8)
        out[(size_t)b * LSEQ * DDIM + (size_t)(l0 + r) * DDIM + (d0 + tx)] = tile[tx][r];
}

// ---------------------------------------------------------------------------
extern "C" void kernel_launch(void* const* d_in, const int* in_sizes, int n_in,
                              void* d_out, int out_size, void* d_ws, size_t ws_size,
                              hipStream_t stream)
{
    (void)in_sizes; (void)n_in; (void)out_size; (void)ws_size;

    const float* x      = (const float*)d_in[0];
    const float* w0     = (const float*)d_in[1];
    const float* b0     = (const float*)d_in[2];
    const float* w1     = (const float*)d_in[3];
    const float* b1     = (const float*)d_in[4];
    const float* w2     = (const float*)d_in[5];
    const float* b2     = (const float*)d_in[6];
    const float* freq   = (const float*)d_in[7];
    const float* deltas = (const float*)d_in[8];
    const float* bias   = (const float*)d_in[9];
    float*       out    = (float*)d_out;

    // workspace layout
    float* kT = (float*)d_ws;                               // D*L
    float* Kf = kT + (size_t)DDIM * LSEQ;                   // D*NFFT*2
    float* xT = Kf + (size_t)DDIM * NFFT * 2;               // B*D*L

    const size_t fft_lds = (size_t)(2 * NFFT + NFFT) * sizeof(float); // re+im+tw

    filter_kernel<<<LSEQ / 16, 32, 0, stream>>>(w0, b0, w1, b1, w2, b2,
                                                freq, deltas, kT);

    kfft_kernel<<<DDIM, 256, fft_lds, stream>>>(kT, Kf);

    dim3 tgrid(LSEQ / 32, DDIM / 32, BDIM);
    dim3 tblk(32, 8);
    transpose_fwd<<<tgrid, tblk, 0, stream>>>(x, xT);

    conv_kernel<<<BDIM * DDIM, 256, fft_lds, stream>>>(xT, Kf, bias);

    transpose_bwd<<<tgrid, tblk, 0, stream>>>(xT, out);
}
